// MultiHeadAttention_73942156968127
// MI455X (gfx1250) — matrix operand order repaired and checked by reading, not compiled, not run on hardware
//
#include <hip/hip_runtime.h>
#include <hip/hip_bf16.h>

#define N_HEADS 8
#define EMB     512
#define BATCH   256
#define NKEYS   2048
#define VTILE   16      // V rows per async stage (16 rows * 2KB = 32KB)

typedef __attribute__((ext_vector_type(16))) __bf16 v16bf;
typedef __attribute__((ext_vector_type(8)))  float  v8f;

// ---- gfx1250 async global->LDS copy support (probe via __has_builtin) ------
#if defined(__has_builtin)
#  if __has_builtin(__builtin_amdgcn_global_load_async_to_lds_b128)
#    define HAVE_ASYNC_COPY 1
#  endif
#  if __has_builtin(__builtin_amdgcn_s_wait_asynccnt)
#    define HAVE_WAIT_ASYNC 1
#  endif
#endif

#if defined(HAVE_ASYNC_COPY)
#  if defined(HAVE_WAIT_ASYNC)
#    define WAIT_ASYNC(n) __builtin_amdgcn_s_wait_asynccnt(n)
#  else
#    define WAIT_ASYNC(n) asm volatile("s_wait_asynccnt %0" ::"i"(n) : "memory")
#  endif

// clang signature: (int4* AS1 src, int4* AS3 dst, imm offset, imm cpol)
typedef int v4i_gcc __attribute__((vector_size(16)));
__device__ __forceinline__ void async_copy_b128(const float* gsrc, float* ldst) {
    __builtin_amdgcn_global_load_async_to_lds_b128(
        (__attribute__((address_space(1))) v4i_gcc*)gsrc,
        (__attribute__((address_space(3))) v4i_gcc*)ldst,
        0, 0);
}
#endif

// ---------------------------------------------------------------------------
// Kernel 1: decode attention, one batch element per workgroup.
// 512 threads = 16 wave32. K streamed direct-to-VGPR (coalesced b128);
// V streamed through a double-buffered async global->LDS pipeline.
// ---------------------------------------------------------------------------
__global__ __launch_bounds__(512)
void attn_decode_kernel(const float* __restrict__ q,
                        const float* __restrict__ K,
                        const float* __restrict__ V,
                        const unsigned char* __restrict__ mask,
                        float* __restrict__ merged)  // [BATCH][EMB]
{
    const int b    = blockIdx.x;
    const int tid  = threadIdx.x;
    const int lane = tid & 31;
    const int wave = tid >> 5;        // 0..15
    const int half = lane >> 4;       // 0 or 1

    __shared__ float s_q[EMB];                  // 2 KB
    __shared__ float s_logits[N_HEADS * NKEYS]; // 64 KB, [h][k]
#if defined(HAVE_ASYNC_COPY)
    __shared__ float s_v[2][VTILE * EMB];       // 2 x 32 KB double buffer
#else
    __shared__ float s_part[4 * EMB];           // 8 KB (fallback reduction)
#endif

    // ---- stage q[b] into LDS -------------------------------------------------
#if defined(HAVE_ASYNC_COPY)
    if (tid < EMB / 4) {  // 128 lanes x b128 = 2KB
        async_copy_b128(q + (size_t)b * EMB + tid * 4, &s_q[tid * 4]);
        WAIT_ASYNC(0);
    }
#else
    s_q[tid] = q[(size_t)b * EMB + tid];
#endif
    __syncthreads();

    const float scale = 0.125f; // 1/sqrt(64)

    // ---- Pass 1: scores. One wave per key; lanes cover the 512 dims with
    // four b128 loads (32 lanes * 16B = 512B contiguous per load).
    for (int k = wave; k < NKEYS; k += 16) {
        const float* krow = K + ((size_t)b * NKEYS + k) * EMB;
        float ps[4];
#pragma unroll
        for (int i = 0; i < 4; ++i) {
            const int d = i * 128 + lane * 4;      // chunk lies inside head 2i+half
            float4 kv = *reinterpret_cast<const float4*>(krow + d);
            ps[i] = kv.x * s_q[d] + kv.y * s_q[d + 1] +
                    kv.z * s_q[d + 2] + kv.w * s_q[d + 3];
        }
#pragma unroll
        for (int off = 1; off < 16; off <<= 1) {   // reduce inside 16-lane halves
#pragma unroll
            for (int i = 0; i < 4; ++i) ps[i] += __shfl_xor(ps[i], off, 32);
        }
        if ((lane & 15) == 0) {
            const bool msk = mask[(size_t)b * NKEYS + k] != 0;
#pragma unroll
            for (int i = 0; i < 4; ++i) {
                const int h = 2 * i + half;
                s_logits[h * NKEYS + k] = msk ? -__builtin_inff() : ps[i] * scale;
            }
        }
    }
    __syncthreads();

    // ---- Pass 2: masked softmax per head (waves 0..7, one head each).
    if (wave < N_HEADS) {
        const int h = wave;
        float mx = -__builtin_inff();
        for (int k = lane; k < NKEYS; k += 32)
            mx = fmaxf(mx, s_logits[h * NKEYS + k]);
#pragma unroll
        for (int off = 1; off < 32; off <<= 1)
            mx = fmaxf(mx, __shfl_xor(mx, off, 32));
        float sum = 0.f;
        for (int k = lane; k < NKEYS; k += 32)
            sum += __expf(s_logits[h * NKEYS + k] - mx);
#pragma unroll
        for (int off = 1; off < 32; off <<= 1)
            sum += __shfl_xor(sum, off, 32);
        const float inv = 1.0f / sum;
        for (int k = lane; k < NKEYS; k += 32)
            s_logits[h * NKEYS + k] = __expf(s_logits[h * NKEYS + k] - mx) * inv;
    }
    __syncthreads();

    // ---- Pass 3: o[d] = sum_k a[h(d)][k] * V[b,k,d] --------------------------
#if defined(HAVE_ASYNC_COPY)
    {
        // Double-buffered async global->LDS pipeline over contiguous V tiles.
        // Each stage: 16 rows * 2KB = 32KB = 2048 b128 chunks; 512 threads
        // issue 4 chunks each -> wait asynccnt<=4 keeps next stage in flight.
        const int h = tid >> 6;       // head of this output dim
        float acc = 0.f;

        auto issue_stage = [&](int buf, int kb) {
            const float* gsrc = V + ((size_t)b * NKEYS + kb) * EMB;
#pragma unroll
            for (int j = 0; j < 4; ++j) {
                const int chunk = tid + j * 512;   // b128 chunk index
                async_copy_b128(gsrc + chunk * 4, &s_v[buf][chunk * 4]);
            }
        };

        issue_stage(0, 0);
        for (int kb = 0; kb < NKEYS; kb += VTILE) {
            const int cur = (kb / VTILE) & 1;
            const bool more = (kb + VTILE) < NKEYS;
            if (more) issue_stage(cur ^ 1, kb + VTILE);
            if (more) { WAIT_ASYNC(4); } else { WAIT_ASYNC(0); }
            __syncthreads();   // ASYNCcnt is per-wave; tile is consumed by all waves
#pragma unroll
            for (int r = 0; r < VTILE; ++r) {
                const float w = s_logits[h * NKEYS + kb + r];   // LDS broadcast per wave
                acc += w * s_v[cur][r * EMB + tid];             // conflict-free b32
            }
            __syncthreads();   // don't let next stage overwrite a tile in use
        }
        merged[(size_t)b * EMB + tid] = acc;   // head-merge order is identity
    }
#else
    {
        // Fallback: direct coalesced float4 streaming, 4 key-chunks x 128 quads.
        const int c  = tid >> 7;
        const int dq = tid & 127;
        const int d0 = dq * 4;
        const int h  = d0 >> 6;
        const int k0 = c * (NKEYS / 4), k1 = k0 + (NKEYS / 4);
        const float* vbase = V + (size_t)b * NKEYS * EMB + d0;
        float ax = 0.f, ay = 0.f, az = 0.f, aw = 0.f;
        for (int k = k0; k < k1; ++k) {
            const float w = s_logits[h * NKEYS + k];
            float4 vv = *reinterpret_cast<const float4*>(vbase + (size_t)k * EMB);
            if (k + 16 < k1)
                __builtin_prefetch(vbase + (size_t)(k + 16) * EMB, 0, 0);
            ax += w * vv.x; ay += w * vv.y; az += w * vv.z; aw += w * vv.w;
        }
        s_part[c * EMB + d0 + 0] = ax;
        s_part[c * EMB + d0 + 1] = ay;
        s_part[c * EMB + d0 + 2] = az;
        s_part[c * EMB + d0 + 3] = aw;
        __syncthreads();
        merged[(size_t)b * EMB + tid] = s_part[tid] + s_part[EMB + tid] +
                                        s_part[2 * EMB + tid] + s_part[3 * EMB + tid];
    }
#endif
}

// ---------------------------------------------------------------------------
// Kernel 2: out[256,512] = merged[256,512] @ W_o^T via v_wmma_f32_16x16x32_bf16.
// One 16x16 output tile per wave; K-loop of 16 WMMAs (K=512), unrolled x2.
// Fragment layouts per CDNA5 ISA 7.12.2 (wave32):
//   A (16x32 bf16): lane m=L%16; half=L/16; VGPR v<4: K=2v+8*half(+1),
//                   v>=4: K=16+2(v-4)+8*half(+1)
//   B (32x16 bf16): lane n=L%16; same K order as A
//   C/D (16x16 f32): lane n=L%16; VGPR r: m = r + 8*half
// ---------------------------------------------------------------------------
__global__ __launch_bounds__(256)
void proj_wmma_kernel(const float* __restrict__ merged,
                      const float* __restrict__ Wo,   // [EMB][EMB], out = merged @ Wo^T
                      float* __restrict__ out)        // [BATCH][EMB]
{
    const int lane = threadIdx.x & 31;
    const int wave = threadIdx.x >> 5;
    const int tile = blockIdx.x * 8 + wave;   // 512 tiles = 16 (M) x 32 (N)
    const int tm   = tile >> 5;               // 0..15
    const int tn   = tile & 31;               // 0..31
    const int half = lane >> 4;
    const int l16  = lane & 15;

    const int row = tm * 16 + l16;  // batch row for A
    const int col = tn * 16 + l16;  // output col for B/D; B[k,n] = Wo[n,k]

    const float* arow = merged + (size_t)row * EMB;
    const float* brow = Wo + (size_t)col * EMB;

    v8f acc = {};
#pragma unroll 2
    for (int kb = 0; kb < EMB; kb += 32) {
        v16bf a, bm;
#pragma unroll
        for (int v = 0; v < 8; ++v) {
            const int ka = kb + 16 * (v >> 2) + 2 * (v & 3) + 8 * half;
            a[2 * v]     = (__bf16)arow[ka];
            a[2 * v + 1] = (__bf16)arow[ka + 1];
            const int kk = kb + 16 * (v >> 2) + 2 * (v & 3) + 8 * half;
            bm[2 * v]     = (__bf16)brow[kk];
            bm[2 * v + 1] = (__bf16)brow[kk + 1];
        }
        acc = __builtin_amdgcn_wmma_f32_16x16x32_bf16(
            /*neg_a=*/false, a, /*neg_b=*/false, bm,
            /*c_mod=*/(short)0, acc, /*reuse_a=*/false, /*reuse_b=*/false);
    }

#pragma unroll
    for (int r = 0; r < 8; ++r) {
        const int m = tm * 16 + r + 8 * half;
        out[(size_t)m * EMB + col] = acc[r];
    }
}

// ---------------------------------------------------------------------------
extern "C" void kernel_launch(void* const* d_in, const int* in_sizes, int n_in,
                              void* d_out, int out_size, void* d_ws, size_t ws_size,
                              hipStream_t stream) {
    (void)in_sizes; (void)n_in; (void)out_size; (void)ws_size;
    const float*         q    = (const float*)d_in[0];          // [256,1,512]
    const float*         K    = (const float*)d_in[1];          // [256,2048,512]
    const float*         V    = (const float*)d_in[2];          // [256,2048,512]
    const float*         Wo   = (const float*)d_in[3];          // [512,512]
    const unsigned char* mask = (const unsigned char*)d_in[4];  // [256,1,2048] bool
    float* out    = (float*)d_out;                              // [256,1,512]
    float* merged = (float*)d_ws;                               // [256,512] scratch

    attn_decode_kernel<<<BATCH, 512, 0, stream>>>(q, K, V, mask, merged);
    proj_wmma_kernel<<<64, 256, 0, stream>>>(merged, Wo, out);
}
